// MPNNLayer_19516331393712
// MI455X (gfx1250) — compile-verified, matrix operand order checked by reading, not run on hardware
//
#include <hip/hip_runtime.h>
#include <hip/hip_bf16.h>

typedef __attribute__((ext_vector_type(16))) __bf16 v16bf;
typedef __attribute__((ext_vector_type(8)))  float  v8f;
typedef __attribute__((ext_vector_type(8)))  unsigned short ushort8;

union Frag16 {
    v16bf   v;
    ushort8 h[2];
};

__device__ __forceinline__ unsigned short f32_to_bf16_bits(float f) {
    __bf16 b = (__bf16)f;                     // native v_cvt_pk_bf16_f32 path
    return __builtin_bit_cast(unsigned short, b);
}

__device__ __forceinline__ unsigned pack_bf16x2(float lo, float hi) {
    return (unsigned)f32_to_bf16_bits(lo) | ((unsigned)f32_to_bf16_bits(hi) << 16);
}

__device__ __forceinline__ float relu1(float x) {
    // v_med3_f32(x, 0, +inf) == clamp to [0, inf): single VALU op
    return __builtin_amdgcn_fmed3f(x, 0.0f, __builtin_inff());
}

// One 16x32 bf16 fragment (ISA 16-bit A layout): lanes 0-15 hold K=k0..k0+7 &
// k0+16..k0+23; lanes 16-31 hold K=k0+8..k0+15 & k0+24..k0+31.
__device__ __forceinline__ void load_frag(Frag16& f, const unsigned short* row, int k0, int half) {
    f.h[0] = *(const ushort8*)(row + k0 +      half * 8);
    f.h[1] = *(const ushort8*)(row + k0 + 16 + half * 8);
}

// ---------------------------------------------------------------- utility kernels

__global__ __launch_bounds__(256) void zero_f32_kernel(float* __restrict__ p, long n) {
    long i = (long)blockIdx.x * blockDim.x + threadIdx.x;
    if (i < n) p[i] = 0.0f;
}

// W: (K x 128) f32 row-major  ->  Wt: (128 x K) bf16 (N-major, contiguous in K)
__global__ __launch_bounds__(256) void cvt_wt_kernel(const float* __restrict__ W,
                                                     unsigned short* __restrict__ Wt, int K) {
    int idx = blockIdx.x * blockDim.x + threadIdx.x;
    if (idx < K * 128) {
        int n = idx / K;
        int k = idx - n * K;
        Wt[n * K + k] = f32_to_bf16_bits(W[k * 128 + n]);
    }
}

// ---------------------------------------------------------------- edge message kernel
// 64 edges per block, 256 threads = 8 waves, wave w owns output columns [16w, 16w+16)

#define TILE_R 64

__global__ __launch_bounds__(256, 1) void edge_msg_kernel(
    const float* __restrict__ h, const int* __restrict__ edges,
    const unsigned short* __restrict__ W1t, const float* __restrict__ b1,
    const unsigned short* __restrict__ W2t, const float* __restrict__ b2,
    float* __restrict__ agg, int n_edges)
{
    __shared__ unsigned short Xs[TILE_R * 256];   // 32 KB: gathered [h_src | h_dst] bf16
    __shared__ unsigned short Hs[TILE_R * 128];   // 16 KB: hidden activations bf16
    __shared__ int src_s[TILE_R];
    __shared__ int dst_s[TILE_R];

    const int tid   = threadIdx.x;
    const int wave  = tid >> 5;
    const int lane  = tid & 31;
    const int half  = lane >> 4;
    const int lanen = lane & 15;
    const int coln  = wave * 16 + lanen;          // 0..127
    const int base  = blockIdx.x * TILE_R;

    if (tid < TILE_R) {
        int e = base + tid;
        if (e < n_edges) { src_s[tid] = edges[2 * e]; dst_s[tid] = edges[2 * e + 1]; }
        else             { src_s[tid] = 0;            dst_s[tid] = -1; }
    }
    __syncthreads();

    // Coalesced gather: float2 per lane, packed bf16x2 store to LDS (b32 ds stores)
    for (int idx = tid; idx < TILE_R * 128; idx += 256) {
        int r  = idx >> 7;            // row in tile
        int cp = idx & 127;           // column-pair index; col = 2*cp
        int node = (cp < 64) ? src_s[r] : dst_s[r];
        if (node < 0) node = 0;
        int cc = (cp & 63) * 2;
        float2 v = *(const float2*)&h[(long)node * 128 + cc];
        *(unsigned*)&Xs[r * 256 + cp * 2] = pack_bf16x2(v.x, v.y);
    }

    // Hoisted weight fragments (VGPR-resident, fed from L2/WGP$)
    Frag16 w1f[8], w2f[4];
    #pragma unroll
    for (int ks = 0; ks < 8; ++ks) load_frag(w1f[ks], W1t + coln * 256, ks * 32, half);
    #pragma unroll
    for (int ks = 0; ks < 4; ++ks) load_frag(w2f[ks], W2t + coln * 128, ks * 32, half);

    const float bias1 = b1[coln];
    const float bias2 = b2[coln];

    __syncthreads();

    for (int st = 0; st < TILE_R / 16; ++st) {
        const int arow = st * 16 + lanen;

        // layer 1: hidden = relu(X @ W1 + b1), K = 256
        Frag16 a1[8];
        #pragma unroll
        for (int ks = 0; ks < 8; ++ks) load_frag(a1[ks], &Xs[arow * 256], ks * 32, half);

        v8f c1;
        #pragma unroll
        for (int j = 0; j < 8; ++j) c1[j] = bias1;
        #pragma unroll
        for (int ks = 0; ks < 8; ++ks)
            c1 = __builtin_amdgcn_wmma_f32_16x16x32_bf16(false, a1[ks].v, false, w1f[ks].v,
                                                         (short)0, c1, false, false);
        #pragma unroll
        for (int j = 0; j < 8; ++j)
            Hs[(st * 16 + j + 8 * half) * 128 + coln] = f32_to_bf16_bits(relu1(c1[j]));
        __syncthreads();

        // layer 2: msg = hidden @ W2 + b2, K = 128
        Frag16 a2[4];
        #pragma unroll
        for (int ks = 0; ks < 4; ++ks) load_frag(a2[ks], &Hs[arow * 128], ks * 32, half);

        v8f c2;
        #pragma unroll
        for (int j = 0; j < 8; ++j) c2[j] = bias2;
        #pragma unroll
        for (int ks = 0; ks < 4; ++ks)
            c2 = __builtin_amdgcn_wmma_f32_16x16x32_bf16(false, a2[ks].v, false, w2f[ks].v,
                                                         (short)0, c2, false, false);

        // scatter-sum into L2-resident accumulator; batch the 8 dst reads as 2x b128
        int4 d0 = *(const int4*)&dst_s[st * 16 + 8 * half];
        int4 d1 = *(const int4*)&dst_s[st * 16 + 8 * half + 4];
        int dn[8] = {d0.x, d0.y, d0.z, d0.w, d1.x, d1.y, d1.z, d1.w};
        #pragma unroll
        for (int j = 0; j < 8; ++j) {
            if (dn[j] >= 0) atomicAdd(&agg[(long)dn[j] * 128 + coln], c2[j]);
        }
        __syncthreads();
    }
}

// ---------------------------------------------------------------- node update kernel
// out = h + relu([h|agg] @ U1 + ub1) @ U2 + ub2

__global__ __launch_bounds__(256, 1) void node_upd_kernel(
    const float* __restrict__ h, const float* __restrict__ agg,
    const unsigned short* __restrict__ W1t, const float* __restrict__ b1,
    const unsigned short* __restrict__ W2t, const float* __restrict__ b2,
    float* __restrict__ out, int n_nodes)
{
    __shared__ unsigned short Xs[TILE_R * 256];
    __shared__ unsigned short Hs[TILE_R * 128];

    const int tid   = threadIdx.x;
    const int wave  = tid >> 5;
    const int lane  = tid & 31;
    const int half  = lane >> 4;
    const int lanen = lane & 15;
    const int coln  = wave * 16 + lanen;
    const int base  = blockIdx.x * TILE_R;

    for (int idx = tid; idx < TILE_R * 128; idx += 256) {
        int r  = idx >> 7;
        int cp = idx & 127;
        long gr = base + r;
        float2 v = make_float2(0.0f, 0.0f);
        if (gr < n_nodes) {
            int cc = (cp & 63) * 2;
            const float* srcp = (cp < 64) ? &h[gr * 128 + cc] : &agg[gr * 128 + cc];
            v = *(const float2*)srcp;
        }
        *(unsigned*)&Xs[r * 256 + cp * 2] = pack_bf16x2(v.x, v.y);
    }

    Frag16 w1f[8], w2f[4];
    #pragma unroll
    for (int ks = 0; ks < 8; ++ks) load_frag(w1f[ks], W1t + coln * 256, ks * 32, half);
    #pragma unroll
    for (int ks = 0; ks < 4; ++ks) load_frag(w2f[ks], W2t + coln * 128, ks * 32, half);

    const float bias1 = b1[coln];
    const float bias2 = b2[coln];

    __syncthreads();

    for (int st = 0; st < TILE_R / 16; ++st) {
        const int arow = st * 16 + lanen;

        Frag16 a1[8];
        #pragma unroll
        for (int ks = 0; ks < 8; ++ks) load_frag(a1[ks], &Xs[arow * 256], ks * 32, half);

        v8f c1;
        #pragma unroll
        for (int j = 0; j < 8; ++j) c1[j] = bias1;
        #pragma unroll
        for (int ks = 0; ks < 8; ++ks)
            c1 = __builtin_amdgcn_wmma_f32_16x16x32_bf16(false, a1[ks].v, false, w1f[ks].v,
                                                         (short)0, c1, false, false);
        #pragma unroll
        for (int j = 0; j < 8; ++j)
            Hs[(st * 16 + j + 8 * half) * 128 + coln] = f32_to_bf16_bits(relu1(c1[j]));
        __syncthreads();

        Frag16 a2[4];
        #pragma unroll
        for (int ks = 0; ks < 4; ++ks) load_frag(a2[ks], &Hs[arow * 128], ks * 32, half);

        v8f c2;
        #pragma unroll
        for (int j = 0; j < 8; ++j) c2[j] = bias2;
        #pragma unroll
        for (int ks = 0; ks < 4; ++ks)
            c2 = __builtin_amdgcn_wmma_f32_16x16x32_bf16(false, a2[ks].v, false, w2f[ks].v,
                                                         (short)0, c2, false, false);

        #pragma unroll
        for (int j = 0; j < 8; ++j) {
            long gr = base + st * 16 + j + 8 * half;
            if (gr < n_nodes) out[gr * 128 + coln] = h[gr * 128 + coln] + c2[j];
        }
        __syncthreads();
    }
}

// ---------------------------------------------------------------- launch

extern "C" void kernel_launch(void* const* d_in, const int* in_sizes, int n_in,
                              void* d_out, int out_size, void* d_ws, size_t ws_size,
                              hipStream_t stream) {
    const float* h      = (const float*)d_in[0];
    const int*   edges  = (const int*)  d_in[1];
    const float* msg_W1 = (const float*)d_in[2];
    const float* msg_b1 = (const float*)d_in[3];
    const float* msg_W2 = (const float*)d_in[4];
    const float* msg_b2 = (const float*)d_in[5];
    const float* upd_W1 = (const float*)d_in[6];
    const float* upd_b1 = (const float*)d_in[7];
    const float* upd_W2 = (const float*)d_in[8];
    const float* upd_b2 = (const float*)d_in[9];
    float* out = (float*)d_out;

    const int n_nodes = in_sizes[0] / 128;
    const int n_edges = in_sizes[1] / 2;

    // workspace layout: [agg f32][msgW1t bf16][msgW2t bf16][updW1t bf16][updW2t bf16]
    char*  ws       = (char*)d_ws;
    size_t aggBytes = ((size_t)n_nodes * 128 * sizeof(float) + 255) & ~(size_t)255;
    float*          agg  = (float*)ws;
    unsigned short* mW1t = (unsigned short*)(ws + aggBytes);
    unsigned short* mW2t = mW1t + 256 * 128;
    unsigned short* uW1t = mW2t + 128 * 128;
    unsigned short* uW2t = uW1t + 256 * 128;

    const long aggElems = (long)n_nodes * 128;
    zero_f32_kernel<<<(unsigned)((aggElems + 255) / 256), 256, 0, stream>>>(agg, aggElems);

    cvt_wt_kernel<<<(256 * 128 + 255) / 256, 256, 0, stream>>>(msg_W1, mW1t, 256);
    cvt_wt_kernel<<<(128 * 128 + 255) / 256, 256, 0, stream>>>(msg_W2, mW2t, 128);
    cvt_wt_kernel<<<(256 * 128 + 255) / 256, 256, 0, stream>>>(upd_W1, uW1t, 256);
    cvt_wt_kernel<<<(128 * 128 + 255) / 256, 256, 0, stream>>>(upd_W2, uW2t, 128);

    edge_msg_kernel<<<(n_edges + TILE_R - 1) / TILE_R, 256, 0, stream>>>(
        h, edges, mW1t, msg_b1, mW2t, msg_b2, agg, n_edges);

    node_upd_kernel<<<(n_nodes + TILE_R - 1) / TILE_R, 256, 0, stream>>>(
        h, agg, uW1t, upd_b1, uW2t, upd_b2, out, n_nodes);
}
